// GCN_19327352832215
// MI455X (gfx1250) — compile-verified, matrix operand order checked by reading, not run on hardware
//
#include <hip/hip_runtime.h>
#include <hip/hip_bf16.h>

// GCN (2x GCNConv + linear head) for MI455X / gfx1250.
// GEMMs via v_wmma_f32_16x16x32_bf16 (f32->bf16 inputs, f32 accum).
// Each wave computes a 32-row x HOUT strip (2 M-tiles) so the 8 weight
// B-fragments per K-step are amortized over 16 WMMAs (1.5 b128 loads/WMMA).
// Edge aggregation via L2-resident float4 gathers + global_atomic_add_f32
// (the 100000x128 feature matrix is 51MB and fits the 192MB L2).

typedef __bf16 bf16_t;
typedef bf16_t v16bf __attribute__((ext_vector_type(16)));
typedef float  v8f   __attribute__((ext_vector_type(8)));

#define IN_DIM 256
#define HID    128
#define NCLS   32

// ---------------- weight transpose + bf16 convert: Wt[n][k] = W[k][n] -------
__global__ void wtrans_kernel(const float* __restrict__ W, bf16_t* __restrict__ Wt,
                              int K, int Hout) {
  int idx = blockIdx.x * blockDim.x + threadIdx.x;
  if (idx >= K * Hout) return;
  int n = idx / K;
  int k = idx - n * K;
  Wt[idx] = (bf16_t)W[k * Hout + n];
}

// ---------------- degree with self loops ------------------------------------
__global__ void deg_init_kernel(float* __restrict__ deg, int n) {
  int i = blockIdx.x * blockDim.x + threadIdx.x;
  if (i < n) deg[i] = 1.0f;  // self loop
}

__global__ void deg_count_kernel(const long long* __restrict__ ei,
                                 float* __restrict__ deg, long long E) {
  long long e = (long long)blockIdx.x * blockDim.x + threadIdx.x;
  if (e < E) atomicAdd(&deg[(int)ei[E + e]], 1.0f);
}

__global__ void dinv_kernel(const float* __restrict__ deg, float* __restrict__ dinv, int n) {
  int i = blockIdx.x * blockDim.x + threadIdx.x;
  if (i < n) {
    float d = deg[i];
    dinv[i] = d > 0.0f ? rsqrtf(d) : 0.0f;
  }
}

// ---------------- WMMA GEMM: C[nrows x HOUT] = A[nrows x K] @ Wt^T ----------
// Wt is [HOUT][K] bf16 (i.e. W transposed). Each wave computes MT 16-row tiles
// over the full HOUT, reusing each B-fragment across all MT tiles per K-step.
// A-frag: lane<16 -> row r, k 0..15 ; lane>=16 -> row r, k 16..31 (per chunk).
// B-frag mirrors with N<->M: each lane loads 16 contiguous bf16 from one Wt row.
template <int K, int HOUT, int MT, bool BIAS>
__global__ __launch_bounds__(256) void gemm_wmma_kernel(
    const float* __restrict__ A, const bf16_t* __restrict__ Wt,
    const float* __restrict__ bias, float* __restrict__ Cmat, int nrows) {
  const int lane = threadIdx.x & 31;
  const int wave = threadIdx.x >> 5;
  const int rowBase = (blockIdx.x * 8 + wave) * (16 * MT);
  if (rowBase >= nrows) return;  // wave-uniform: EXEC stays all-ones for WMMA

  const int half = lane >> 4;  // 0: k 0..15 / rows 0..7 ; 1: k 16..31 / rows 8..15
  const int r    = lane & 15;

  constexpr int NT = HOUT / 16;
  v8f acc[MT][NT] = {};

  // Per-tile A row pointers; inactive tiles clamp to row 0 (loads stay valid,
  // results are discarded at store time). All conditions are wave-uniform.
  bool act[MT];
  const float* aBase[MT];
#pragma unroll
  for (int t = 0; t < MT; ++t) {
    const int row = rowBase + t * 16;
    act[t] = row < nrows;
    aBase[t] = A + (size_t)((act[t] ? row : 0) + r) * K + half * 16;
  }

  for (int kt = 0; kt < K; kt += 32) {
    // Load 16 consecutive f32 per tile of this lane's A row chunk -> bf16.
    v16bf a[MT];
#pragma unroll
    for (int t = 0; t < MT; ++t) {
      const float4* ap = (const float4*)(aBase[t] + kt);
#pragma unroll
      for (int q = 0; q < 4; ++q) {
        float4 f = ap[q];
        a[t][4 * q + 0] = (bf16_t)f.x;
        a[t][4 * q + 1] = (bf16_t)f.y;
        a[t][4 * q + 2] = (bf16_t)f.z;
        a[t][4 * q + 3] = (bf16_t)f.w;
      }
    }
#pragma unroll
    for (int nt = 0; nt < NT; ++nt) {
      const v16bf b =
          *(const v16bf*)(Wt + (size_t)(nt * 16 + r) * K + half * 16 + kt);
#pragma unroll
      for (int t = 0; t < MT; ++t) {
        acc[t][nt] = __builtin_amdgcn_wmma_f32_16x16x32_bf16(
            /*neg_a=*/false, a[t], /*neg_b=*/false, b,
            /*c_mod=*/(short)0, acc[t][nt], /*reuse_a=*/false, /*reuse_b=*/false);
      }
    }
  }

  // C layout: vgpr rr -> row half*8+rr, lane -> column r (within each 16-col tile)
#pragma unroll
  for (int t = 0; t < MT; ++t) {
    if (!act[t]) continue;
#pragma unroll
    for (int nt = 0; nt < NT; ++nt) {
      const int n = nt * 16 + r;
      const float bval = BIAS ? bias[n] : 0.0f;
#pragma unroll
      for (int rr = 0; rr < 8; ++rr) {
        const int m = rowBase + t * 16 + half * 8 + rr;
        Cmat[(size_t)m * HOUT + n] = acc[t][nt][rr] + bval;
      }
    }
  }
}

// ---------------- aggregation (HID = 128 fixed) -----------------------------
// Self-loop term; also fully initializes H (workspace is poisoned by harness).
__global__ void agg_self_kernel(const float* __restrict__ Y, const float* __restrict__ dinv,
                                float* __restrict__ H, int n) {
  size_t idx = (size_t)blockIdx.x * blockDim.x + threadIdx.x;  // n*32 float4 chunks
  int node = (int)(idx >> 5);
  if (node >= n) return;
  int c = (int)(idx & 31) * 4;
  float s = dinv[node];
  s = s * s;
  float4 y = *(const float4*)(Y + (size_t)node * HID + c);
  float4 h;
  h.x = y.x * s; h.y = y.y * s; h.z = y.z * s; h.w = y.w * s;
  *(float4*)(H + (size_t)node * HID + c) = h;
}

// One wave per edge: 32 lanes x float4 = 128 features. src/dst wave-uniform.
__global__ void agg_edges_kernel(const long long* __restrict__ ei,
                                 const float* __restrict__ Y,
                                 const float* __restrict__ dinv,
                                 float* __restrict__ H, long long E) {
  long long idx = (long long)blockIdx.x * blockDim.x + threadIdx.x;
  long long e = idx >> 5;
  if (e >= E) return;
  int s = (int)ei[e];
  int d = (int)ei[E + e];
  float norm = dinv[s] * dinv[d];
  int c = (int)(idx & 31) * 4;
  float4 y = *(const float4*)(Y + (size_t)s * HID + c);
  float* hp = H + (size_t)d * HID + c;
  atomicAdd(hp + 0, y.x * norm);
  atomicAdd(hp + 1, y.y * norm);
  atomicAdd(hp + 2, y.z * norm);
  atomicAdd(hp + 3, y.w * norm);
}

__global__ void bias_relu_kernel(float* __restrict__ H, const float* __restrict__ b, int n) {
  size_t idx = (size_t)blockIdx.x * blockDim.x + threadIdx.x;  // n*32 float4 chunks
  int node = (int)(idx >> 5);
  if (node >= n) return;
  int c = (int)(idx & 31) * 4;
  float* hp = H + (size_t)node * HID + c;
  float4 h = *(float4*)hp;
  h.x = fmaxf(h.x + b[c + 0], 0.0f);
  h.y = fmaxf(h.y + b[c + 1], 0.0f);
  h.z = fmaxf(h.z + b[c + 2], 0.0f);
  h.w = fmaxf(h.w + b[c + 3], 0.0f);
  *(float4*)hp = h;
}

// ---------------- launch ----------------------------------------------------
extern "C" void kernel_launch(void* const* d_in, const int* in_sizes, int n_in,
                              void* d_out, int out_size, void* d_ws, size_t ws_size,
                              hipStream_t stream) {
  const float*     X  = (const float*)d_in[0];
  const long long* ei = (const long long*)d_in[1];  // int64 [2][E]
  const float*     W1 = (const float*)d_in[2];
  const float*     b1 = (const float*)d_in[3];
  const float*     W2 = (const float*)d_in[4];
  const float*     b2 = (const float*)d_in[5];
  const float*     Wc = (const float*)d_in[6];
  const float*     bc = (const float*)d_in[7];
  float*           out = (float*)d_out;

  const int       n = in_sizes[0] / IN_DIM;  // 100000
  const long long E = (long long)in_sizes[1] / 2;

  // workspace carve-out (256B aligned)
  char* ws = (char*)d_ws;
  size_t off = 0;
  auto take = [&](size_t bytes) -> void* {
    void* p = ws + off;
    off += (bytes + 255) & ~(size_t)255;
    return p;
  };
  float*  deg  = (float*)take((size_t)n * 4);
  float*  dinv = (float*)take((size_t)n * 4);
  float*  Y    = (float*)take((size_t)n * HID * 4);  // GEMM output
  float*  H    = (float*)take((size_t)n * HID * 4);  // aggregated activation
  bf16_t* W1t  = (bf16_t*)take((size_t)HID * IN_DIM * 2);
  bf16_t* W2t  = (bf16_t*)take((size_t)HID * HID * 2);
  bf16_t* Wct  = (bf16_t*)take((size_t)NCLS * HID * 2);

  const int TB = 256;
  dim3 blk(TB);

  // weights -> transposed bf16
  wtrans_kernel<<<(IN_DIM * HID + TB - 1) / TB, blk, 0, stream>>>(W1, W1t, IN_DIM, HID);
  wtrans_kernel<<<(HID * HID + TB - 1) / TB, blk, 0, stream>>>(W2, W2t, HID, HID);
  wtrans_kernel<<<(HID * NCLS + TB - 1) / TB, blk, 0, stream>>>(Wc, Wct, HID, NCLS);

  // degrees (self loops included), dinv = rsqrt(deg)
  deg_init_kernel<<<(n + TB - 1) / TB, blk, 0, stream>>>(deg, n);
  deg_count_kernel<<<(unsigned)((E + TB - 1) / TB), blk, 0, stream>>>(ei, deg, E);
  dinv_kernel<<<(n + TB - 1) / TB, blk, 0, stream>>>(deg, dinv, n);

  const unsigned gemmGrid = (unsigned)((n + 255) / 256);  // 8 waves x 32 rows
  const unsigned nodeGrid = (unsigned)(((size_t)n * 32 + TB - 1) / TB);
  const unsigned edgeGrid = (unsigned)((E * 32 + TB - 1) / TB);

  // ---- layer 1: Y = X @ W1 ; H = scatter(norm * Y) ; H = relu(H + b1)
  gemm_wmma_kernel<IN_DIM, HID, 2, false><<<gemmGrid, blk, 0, stream>>>(X, W1t, nullptr, Y, n);
  agg_self_kernel<<<nodeGrid, blk, 0, stream>>>(Y, dinv, H, n);
  agg_edges_kernel<<<edgeGrid, blk, 0, stream>>>(ei, Y, dinv, H, E);
  bias_relu_kernel<<<nodeGrid, blk, 0, stream>>>(H, b1, n);

  // ---- layer 2
  gemm_wmma_kernel<HID, HID, 2, false><<<gemmGrid, blk, 0, stream>>>(H, W2t, nullptr, Y, n);
  agg_self_kernel<<<nodeGrid, blk, 0, stream>>>(Y, dinv, H, n);
  agg_edges_kernel<<<edgeGrid, blk, 0, stream>>>(ei, Y, dinv, H, E);
  bias_relu_kernel<<<nodeGrid, blk, 0, stream>>>(H, b2, n);

  // ---- classifier: out = H @ Wc + bc
  gemm_wmma_kernel<HID, NCLS, 2, true><<<gemmGrid, blk, 0, stream>>>(H, Wct, bc, out, n);
}
